// MIL_75720273429146
// MI455X (gfx1250) — matrix operand order repaired and checked by reading, not compiled, not run on hardware
//
#include <hip/hip_runtime.h>
#include <hip/hip_bf16.h>

typedef __attribute__((ext_vector_type(16))) _Float16 v16h;
typedef __attribute__((ext_vector_type(8)))  float    v8f;
typedef __attribute__((ext_vector_type(4)))  unsigned int v4u;
typedef __attribute__((ext_vector_type(8)))  int      v8i;
typedef __attribute__((ext_vector_type(4)))  int      v4i;

#define T_LEN   2048
#define D_DIM   1024
#define H1_DIM  512
#define H2_DIM  32
#define M_ROWS  (64 * T_LEN)     // 131072
#define TILE_M  128              // rows per workgroup

#define HBUF_STRIDE 520
#define HBUF_BYTES  (128 * HBUF_STRIDE * 2)   // 133120
#define BSTAGE_BYTES 8192                     // one 32(K) x 128(N) f16 panel
#define SMEM_BYTES  (HBUF_BYTES + 2 * BSTAGE_BYTES)  // 149504

// ---------------------------------------------------------------------------
// Prep: convert W1 (1024x512) and W2 (512x32) fp32 -> f16, pre-swizzled into
// WMMA B-fragment layout: tile = 32(K) x 16(N); lane l holds column n = l&15,
// K-range kb..kb+15 where kb = (l<16 ? 0 : 16); the 16 halves per lane are
// stored contiguously so fragments load as one v16h (2x b128).
// ---------------------------------------------------------------------------
__global__ __launch_bounds__(256) void MIL_prep_weights(
    const float* __restrict__ W1, const float* __restrict__ W2,
    _Float16* __restrict__ W1h, _Float16* __restrict__ W2h)
{
    int gid = blockIdx.x * 256 + threadIdx.x;
    if (gid < 32768) {                       // W1: 32 ktiles * 32 ntiles * 32 lanes
        int tile = gid >> 5, lane = gid & 31;
        int kt = tile >> 5, nt = tile & 31;
        int n = lane & 15, kb = (lane < 16) ? 0 : 16;
        _Float16* dst = W1h + (size_t)gid * 16;
        const float* src = W1 + (size_t)(kt * 32 + kb) * H1_DIM + nt * 16 + n;
        #pragma unroll
        for (int i = 0; i < 16; ++i) dst[i] = (_Float16)src[(size_t)i * H1_DIM];
    } else if (gid < 32768 + 1024) {         // W2: 16 ktiles * 2 ntiles * 32 lanes
        int g2 = gid - 32768;
        int tile = g2 >> 5, lane = g2 & 31;
        int kt = tile >> 1, nt = tile & 1;
        int n = lane & 15, kb = (lane < 16) ? 0 : 16;
        _Float16* dst = W2h + (size_t)g2 * 16;
        const float* src = W2 + (size_t)(kt * 32 + kb) * H2_DIM + nt * 16 + n;
        #pragma unroll
        for (int i = 0; i < 16; ++i) dst[i] = (_Float16)src[(size_t)i * H2_DIM];
    }
}

// ---------------------------------------------------------------------------
// TDM: DMA one contiguous 8KB fragment panel (global -> LDS).
// D# group0: count=1, lds_addr, 57-bit global addr, type=2.
// D# group1: data_size=3 (8B elems), tensor_dim0=tile_dim0=1024, 1 row.
// 6-arg builtin (clang-23 / therock headers): (g0, g1, g2, g3, g4, cpol)
// ---------------------------------------------------------------------------
__device__ __forceinline__ void tdm_load_panel(const void* gsrc, unsigned lds_off)
{
    unsigned long long ga = (unsigned long long)(uintptr_t)gsrc;
    v4u g0;
    g0[0] = 1u;                                   // count=1, no gather
    g0[1] = lds_off;                              // lds_addr (bytes)
    g0[2] = (unsigned)(ga & 0xFFFFFFFFu);         // global_addr[31:0]
    g0[3] = (unsigned)((ga >> 32) & 0x01FFFFFFu)  // global_addr[56:32]
          | (2u << 30);                           // type=2 ("image")
    v8i g1;
    g1[0] = (int)(3u << 16);                      // data_size=3 -> 8-byte elements
    g1[1] = (int)(1024u << 16);                   // tensor_dim0[15:0]=1024
    g1[2] = (int)(1u << 16);                      // tensor_dim0[31:16]=0, tensor_dim1=1
    g1[3] = (int)(1024u << 16);                   // tile_dim0=1024 (8KB)
    g1[4] = 0;                                    // tile_dim1=0, tile_dim2=0 (unused)
    g1[5] = 1024;                                 // tensor_dim0_stride[31:0]
    g1[6] = 0;
    g1[7] = 0;
    v4i z4 = {0, 0, 0, 0};
    v8i z8 = {0, 0, 0, 0, 0, 0, 0, 0};
    __builtin_amdgcn_tensor_load_to_lds(g0, g1, z4, z4, z8, 0);
}

// ---------------------------------------------------------------------------
// Fused: h = relu(X*W1+b1) (to LDS f16) ; h2 = h*W2+b2 ; logits = sigmoid(h2.W3+b3)
// 8 waves / 256 threads per WG; each wave owns a 16-row strip of a 128-row tile.
// W1 fragment panels are TDM-staged into double-buffered LDS, shared by waves.
// ---------------------------------------------------------------------------
__global__ __launch_bounds__(256) void MIL_fused_mlp(
    const float* __restrict__ X, const _Float16* __restrict__ W1h,
    const float* __restrict__ b1, const _Float16* __restrict__ W2h,
    const float* __restrict__ b2, const float* __restrict__ W3,
    const float* __restrict__ b3, float* __restrict__ logits)
{
    extern __shared__ char smem[];
    _Float16 (*hbuf)[HBUF_STRIDE] = (_Float16 (*)[HBUF_STRIDE])smem;
    _Float16* bstage0 = (_Float16*)(smem + HBUF_BYTES);
    _Float16* bstage1 = (_Float16*)(smem + HBUF_BYTES + BSTAGE_BYTES);
    // phase-2 scratch aliases the (then idle) staging buffers
    float (*h2buf)[H2_DIM] = (float (*)[H2_DIM])(smem + HBUF_BYTES);

    const int tid  = threadIdx.x;
    const int w    = tid >> 5;              // wave 0..7
    const int lane = tid & 31;
    const int m    = lane & 15;
    const int kbA  = (lane < 16) ? 0 : 16;  // A-fragment K base (contiguous K order, matches B)
    const int sr   = w * 16;                // strip row within WG tile
    const int row0 = blockIdx.x * TILE_M;

    const float* Arow = X + (size_t)(row0 + sr + m) * D_DIM;
    const unsigned lds_b0 = (unsigned)(uintptr_t)(void*)bstage0;
    const unsigned lds_b1 = (unsigned)(uintptr_t)(void*)bstage1;

    // ---------------- Phase 1: GEMM1 (K=1024) + bias + relu -> LDS f16 ----
    for (int nc = 0; nc < 4; ++nc) {
        const int n0 = nc * 128;
        const int nb = n0 >> 4;             // first 16-col tile of this chunk
        // preload panel for k-step 0
        if (w == 0)
            tdm_load_panel(W1h + (size_t)(0 * 32 + nb) * 512, lds_b0);

        v8f acc[8] = {};
        for (int ks = 0; ks < 32; ++ks) {
            if (w == 0) __builtin_amdgcn_s_wait_tensorcnt(0);
            __syncthreads();                // panel[ks&1] ready; panel[(ks+1)&1] idle
            if (w == 0 && ks < 31)
                tdm_load_panel(W1h + (size_t)((ks + 1) * 32 + nb) * 512,
                               (ks & 1) ? lds_b0 : lds_b1);

            const int k0 = ks * 32;
            // A fragment: lane<16 holds K=k0+[0..15]; lane>=16: [16..31]
            const float* ap = Arow + k0 + kbA;
            float4 x0 = *(const float4*)(ap + 0);
            float4 x1 = *(const float4*)(ap + 4);
            float4 x2 = *(const float4*)(ap + 8);
            float4 x3 = *(const float4*)(ap + 12);
            v16h af;
            af[0]  = (_Float16)x0.x; af[1]  = (_Float16)x0.y;
            af[2]  = (_Float16)x0.z; af[3]  = (_Float16)x0.w;
            af[4]  = (_Float16)x1.x; af[5]  = (_Float16)x1.y;
            af[6]  = (_Float16)x1.z; af[7]  = (_Float16)x1.w;
            af[8]  = (_Float16)x2.x; af[9]  = (_Float16)x2.y;
            af[10] = (_Float16)x2.z; af[11] = (_Float16)x2.w;
            af[12] = (_Float16)x3.x; af[13] = (_Float16)x3.y;
            af[14] = (_Float16)x3.z; af[15] = (_Float16)x3.w;

            const _Float16* bc = (ks & 1) ? bstage1 : bstage0;
            #pragma unroll
            for (int nt = 0; nt < 8; ++nt) {
                v16h bf = *(const v16h*)(bc + (nt * 32 + lane) * 16);
                acc[nt] = __builtin_amdgcn_wmma_f32_16x16x32_f16(
                    false, af, false, bf, (short)0, acc[nt], false, false);
            }
        }
        // epilogue: bias + relu, scatter C layout -> row-major LDS f16
        #pragma unroll
        for (int nt = 0; nt < 8; ++nt) {
            const int n  = n0 + nt * 16 + m;
            const float bv = b1[n];
            #pragma unroll
            for (int j = 0; j < 8; ++j) {
                float v = acc[nt][j] + bv;
                v = v > 0.0f ? v : 0.0f;
                const int row = sr + j + ((lane < 16) ? 0 : 8);
                hbuf[row][n] = (_Float16)v;
            }
        }
    }
    __syncthreads();                        // hbuf complete; staging now reusable

    // ---------------- Phase 2: GEMM2 (K=512, N=32) out of LDS ------------
    v8f acc2[2] = {};
    for (int k0 = 0; k0 < H1_DIM; k0 += 32) {
        const _Float16* hp = &hbuf[sr + m][k0 + kbA];
        v16h af;
        #pragma unroll
        for (int i = 0; i < 8; ++i) { af[i] = hp[i]; af[8 + i] = hp[8 + i]; }
        const int ktile = k0 >> 5;
        #pragma unroll
        for (int ni = 0; ni < 2; ++ni) {
            v16h bf = *(const v16h*)(W2h + ((size_t)((ktile * 2 + ni) * 32 + lane)) * 16);
            acc2[ni] = __builtin_amdgcn_wmma_f32_16x16x32_f16(
                false, af, false, bf, (short)0, acc2[ni], false, false);
        }
    }
    #pragma unroll
    for (int ni = 0; ni < 2; ++ni) {
        const int n = ni * 16 + m;
        const float bv = b2[n];
        #pragma unroll
        for (int j = 0; j < 8; ++j) {
            const int row = sr + j + ((lane < 16) ? 0 : 8);
            h2buf[row][n] = acc2[ni][j] + bv;
        }
    }
    __syncthreads();

    // ---------------- Phase 3: dot with W3 + sigmoid -> logits -----------
    if (tid < TILE_M) {
        float s = b3[0];
        #pragma unroll
        for (int n = 0; n < H2_DIM; ++n) s += h2buf[tid][n] * W3[n];
        logits[row0 + tid] = 1.0f / (1.0f + __expf(-s));
    }
}

// ---------------------------------------------------------------------------
// Top-k mean per batch row. Exact k-th largest via integer bisection on the
// float bit pattern (all candidates >= 0, so uint order == float order).
// ---------------------------------------------------------------------------
__global__ __launch_bounds__(256) void MIL_topk_mean(
    const float* __restrict__ logits, const int* __restrict__ seq_len,
    float* __restrict__ out)
{
    __shared__ int   scnt[256];
    __shared__ float ssum[256];
    const int b = blockIdx.x, tid = threadIdx.x;
    const int L = seq_len[b];
    const int k = L / 16 + 1;            // 2 <= k <= 129, k <= L

    unsigned bits[8]; float val[8];
    #pragma unroll
    for (int i = 0; i < 8; ++i) {
        const int pos = tid + i * 256;
        const float v = logits[(size_t)b * T_LEN + pos];
        const bool ok = pos < L;
        val[i]  = ok ? v : 0.0f;
        bits[i] = ok ? __float_as_uint(v) : 0u;   // sigmoid > 0, masked=0 sorts last
    }

    unsigned lo = 0u, hi = 0x3F800000u;  // sigmoid <= 1.0
    for (int it = 0; it < 31 && lo < hi; ++it) {
        const unsigned mid = lo + ((hi - lo + 1u) >> 1);
        int c = 0;
        #pragma unroll
        for (int i = 0; i < 8; ++i) c += (bits[i] >= mid);
        scnt[tid] = c; __syncthreads();
        for (int s = 128; s > 0; s >>= 1) {
            if (tid < s) scnt[tid] += scnt[tid + s];
            __syncthreads();
        }
        const int total = scnt[0]; __syncthreads();
        if (total >= k) lo = mid; else hi = mid - 1u;
    }
    const unsigned piv = lo;             // exact bits of k-th largest
    const float pv = __uint_as_float(piv);

    int c = 0; float s = 0.0f;
    #pragma unroll
    for (int i = 0; i < 8; ++i) { if (bits[i] > piv) { c++; s += val[i]; } }
    scnt[tid] = c; ssum[tid] = s; __syncthreads();
    for (int st = 128; st > 0; st >>= 1) {
        if (tid < st) { scnt[tid] += scnt[tid + st]; ssum[tid] += ssum[tid + st]; }
        __syncthreads();
    }
    if (tid == 0) {
        const float total = ssum[0] + (float)(k - scnt[0]) * pv;
        out[b] = total / (float)k;
    }
}

// ---------------------------------------------------------------------------
extern "C" void kernel_launch(void* const* d_in, const int* in_sizes, int n_in,
                              void* d_out, int out_size, void* d_ws, size_t ws_size,
                              hipStream_t stream) {
    const float* X   = (const float*)d_in[0];
    const int*   seq = (const int*)  d_in[1];
    const float* W1  = (const float*)d_in[2];
    const float* b1  = (const float*)d_in[3];
    const float* W2  = (const float*)d_in[4];
    const float* b2  = (const float*)d_in[5];
    const float* W3  = (const float*)d_in[6];
    const float* b3  = (const float*)d_in[7];
    float* out = (float*)d_out;

    // workspace layout
    char* ws = (char*)d_ws;
    float*    logits = (float*)ws;                                   // 512 KB
    _Float16* W1h    = (_Float16*)(ws + (size_t)M_ROWS * 4);         // 1 MB
    _Float16* W2h    = (_Float16*)(ws + (size_t)M_ROWS * 4 + (size_t)D_DIM * H1_DIM * 2);

    (void)hipFuncSetAttribute((const void*)MIL_fused_mlp,
                              hipFuncAttributeMaxDynamicSharedMemorySize,
                              (int)SMEM_BYTES);

    MIL_prep_weights<<<132, 256, 0, stream>>>(W1, W2, W1h, W2h);
    MIL_fused_mlp<<<M_ROWS / TILE_M, 256, SMEM_BYTES, stream>>>(
        X, W1h, b1, W2h, b2, W3, b3, logits);
    MIL_topk_mean<<<64, 256, 0, stream>>>(logits, seq, out);
}
